// JetBlock_24043226923595
// MI455X (gfx1250) — matrix operand order repaired and checked by reading, not compiled, not run on hardware
//
#include <hip/hip_runtime.h>
#include <stdint.h>

// ---------------- problem constants ----------------
constexpr int kT   = 2048;
constexpr int kHID = 1536;
constexpr int kH   = 12;
constexpr int kDK  = 128;
constexpr int kDV  = 128;
constexpr int kKEY = kH * kDK;       // 1536
constexpr int kVAL = kH * kDV;       // 1536
constexpr int kCONV = 4;
constexpr int kGMID = kHID / 8;      // 192
constexpr int kWCOLS = kVAL * kCONV; // 6144
constexpr int kCHUNK = 32;
#define SCALE_QK 0.08838834764831845f  // DK^-0.5

typedef __bf16 bf16;
typedef __attribute__((ext_vector_type(16))) __bf16 v16bf;
typedef __attribute__((ext_vector_type(8)))  float  v8f;

// ---------------- helpers ----------------
__device__ __forceinline__ bf16 f2bf(float f) { return (bf16)f; }   // native cvt (RNE)
__device__ __forceinline__ float bf2f(bf16 b) { return (float)b; }
__device__ __forceinline__ float siluf(float x) { return x / (1.f + __expf(-x)); }
__device__ __forceinline__ v8f vzero8() {
  v8f z;
#pragma unroll
  for (int i = 0; i < 8; ++i) z[i] = 0.f;
  return z;
}
__device__ __forceinline__ v8f wmma_bf16(v16bf a, v16bf b, v8f c) {
  return __builtin_amdgcn_wmma_f32_16x16x32_bf16(false, a, false, b, (short)0, c,
                                                 false, false);
}

// async copy 16B global -> LDS (CDNA5 GLOBAL_LOAD_ASYNC_TO_LDS_B128, ASYNCcnt)
__device__ __forceinline__ void async_b128(unsigned lds_off, const void* gptr) {
  asm volatile("global_load_async_to_lds_b128 %0, %1, off" ::"v"(lds_off), "v"(gptr)
               : "memory");
}
__device__ __forceinline__ void wait_async0() {
#if __has_builtin(__builtin_amdgcn_s_wait_asynccnt)
  __builtin_amdgcn_s_wait_asynccnt(0);
#else
  asm volatile("s_wait_asynccnt 0" ::: "memory");
#endif
}
__device__ __forceinline__ unsigned lds_addr32(const void* p) {
  return (unsigned)(unsigned long long)p;  // low 32 bits of generic addr == LDS offset
}

// A fragment: 16x32 bf16 tile, element A[m][k] at p[m*rm + k*rk]
__device__ __forceinline__ v16bf load_a(const bf16* p, int rm, int rk, int lane) {
  const int m = lane & 15, half = lane >> 4;
  v16bf a;
#pragma unroll
  for (int i = 0; i < 8; ++i) {
    const int kb = ((i >> 2) << 4) + (half << 3) + ((i & 3) << 1);
    a[2 * i]     = p[m * rm + kb * rk];
    a[2 * i + 1] = p[m * rm + (kb + 1) * rk];
  }
  return a;
}
// B fragment: 32x16 bf16 tile, element B[k][n] at p[k*rs + n*cs]
__device__ __forceinline__ v16bf load_b(const bf16* p, int rs, int cs, int lane) {
  const int n = lane & 15, kh = (lane >> 4) << 3;
  v16bf b;
#pragma unroll
  for (int e = 0; e < 16; ++e) b[e] = p[(((e >> 3) << 4) + kh + (e & 7)) * rs + n * cs];
  return b;
}

__device__ __forceinline__ void store_out(float* p, float v) { *p = v; }
__device__ __forceinline__ void store_out(bf16* p, float v) { *p = f2bf(v); }

// ---------------- f32 -> bf16 convert (x4 vectorized) ----------------
__global__ void cvt_kernel(const float* __restrict__ in, bf16* __restrict__ out,
                           int n4) {
  int i = blockIdx.x * blockDim.x + threadIdx.x;
  if (i < n4) {
    float4 f = ((const float4*)in)[i];
    bf16 t[4] = {f2bf(f.x), f2bf(f.y), f2bf(f.z), f2bf(f.w)};
    ((uint2*)out)[i] = *(const uint2*)t;
  }
}

// ---------------- 128x128 WMMA GEMM, async-LDS double buffered ----------------
// C[M,N] = act(A[M,K] @ B[K,N]); M%128==0, N%128==0, K%32==0.
template <int ACT, typename OUT>
__global__ __launch_bounds__(256) void gemm128_kernel(const bf16* __restrict__ A,
                                                      const bf16* __restrict__ B,
                                                      OUT* __restrict__ C, int M,
                                                      int N, int K) {
  __shared__ alignas(16) bf16 As[2][128][32];
  __shared__ alignas(16) bf16 Bt[2][128][32];  // transposed: [n][k]
  const int tid = threadIdx.x, lane = tid & 31, wave = tid >> 5;
  const int ln = lane & 15, lh = lane >> 4;
  const int wm = wave >> 2, wn = wave & 3;
  const int m0 = blockIdx.y * 128, n0 = blockIdx.x * 128;

  v8f acc[4][2];
#pragma unroll
  for (int mi = 0; mi < 4; ++mi)
#pragma unroll
    for (int ni = 0; ni < 2; ++ni) acc[mi][ni] = vzero8();

  const int rowA = tid >> 1, colA = (tid & 1) << 4;   // 16 bf16 per thread
  const int rowB = tid >> 3, colB = (tid & 7) << 4;   // 16 bf16 per thread

  uint4 rb0, rb1;
  {  // prologue: async A slab 0, B regs slab 0
    const bf16* g = &A[(size_t)(m0 + rowA) * K + colA];
    unsigned l = lds_addr32(&As[0][rowA][colA]);
    async_b128(l, g);
    async_b128(l + 16, g + 8);
    const bf16* gb = &B[(size_t)rowB * N + n0 + colB];
    rb0 = *(const uint4*)gb;
    rb1 = *(const uint4*)(gb + 8);
  }

  const int nk = K / 32;
  for (int it = 0; it < nk; ++it) {
    const int cur = it & 1;
    {  // transpose-store B regs into Bt[cur]
      const bf16* p0 = (const bf16*)&rb0;
      const bf16* p1 = (const bf16*)&rb1;
#pragma unroll
      for (int e = 0; e < 8; ++e) {
        Bt[cur][colB + e][rowB]     = p0[e];
        Bt[cur][colB + 8 + e][rowB] = p1[e];
      }
    }
    wait_async0();
    __syncthreads();
    if (it + 1 < nk) {
      const int k0 = (it + 1) * 32;
      const bf16* g = &A[(size_t)(m0 + rowA) * K + k0 + colA];
      unsigned l = lds_addr32(&As[cur ^ 1][rowA][colA]);
      async_b128(l, g);
      async_b128(l + 16, g + 8);
      const bf16* gb = &B[(size_t)(k0 + rowB) * N + n0 + colB];
      __builtin_prefetch(gb, 0, 1);
      rb0 = *(const uint4*)gb;
      rb1 = *(const uint4*)(gb + 8);
    }
    v16bf afr[4], bfr[2];
#pragma unroll
    for (int mi = 0; mi < 4; ++mi)
      afr[mi] = load_a(&As[cur][wm * 64 + mi * 16][0], 32, 1, lane);
#pragma unroll
    for (int ni = 0; ni < 2; ++ni)
      bfr[ni] = load_b(&Bt[cur][wn * 32 + ni * 16][0], 1, 32, lane);
#pragma unroll
    for (int mi = 0; mi < 4; ++mi)
#pragma unroll
      for (int ni = 0; ni < 2; ++ni)
        acc[mi][ni] = wmma_bf16(afr[mi], bfr[ni], acc[mi][ni]);
  }

#pragma unroll
  for (int mi = 0; mi < 4; ++mi)
#pragma unroll
    for (int ni = 0; ni < 2; ++ni)
#pragma unroll
      for (int i = 0; i < 8; ++i) {
        const int m = m0 + wm * 64 + mi * 16 + lh * 8 + i;
        const int n = n0 + wn * 32 + ni * 16 + ln;
        float v = acc[mi][ni][i];
        if (ACT == 1) v = siluf(v);
        store_out(&C[(size_t)m * N + n], v);
      }
}

// ---------------- 64x64 WMMA GEMM (for N=192) ----------------
template <int ACT, typename OUT>
__global__ __launch_bounds__(256) void gemm64_kernel(const bf16* __restrict__ A,
                                                     const bf16* __restrict__ B,
                                                     OUT* __restrict__ C, int M, int N,
                                                     int K) {
  __shared__ alignas(16) bf16 As[64][32];
  __shared__ alignas(16) bf16 Bts[64][32];  // transposed [n][k]
  const int tid = threadIdx.x, lane = tid & 31, wave = tid >> 5;
  const int ln = lane & 15, lh = lane >> 4;
  const int m0 = blockIdx.y * 64, n0 = blockIdx.x * 64;
  v8f acc[2];
  acc[0] = vzero8();
  acc[1] = vzero8();
  const int ra = tid >> 2, ca = (tid & 3) << 3;
  const int rb = tid >> 3, cb = (tid & 7) << 3;
  for (int k0 = 0; k0 < K; k0 += 32) {
    *(uint4*)&As[ra][ca] = *(const uint4*)&A[(size_t)(m0 + ra) * K + k0 + ca];
    uint4 bx = *(const uint4*)&B[(size_t)(k0 + rb) * N + n0 + cb];
    const bf16* bp = (const bf16*)&bx;
#pragma unroll
    for (int e = 0; e < 8; ++e) Bts[cb + e][rb] = bp[e];
    if (k0 + 32 < K) {
      __builtin_prefetch(&A[(size_t)(m0 + ra) * K + k0 + 32 + ca], 0, 1);
      __builtin_prefetch(&B[(size_t)(k0 + 32 + rb) * N + n0 + cb], 0, 1);
    }
    __syncthreads();
#pragma unroll
    for (int u = 0; u < 2; ++u) {
      const int ti = wave * 2 + u, tr = ti >> 2, tc = ti & 3;
      v16bf a = load_a(&As[tr * 16][0], 32, 1, lane);
      v16bf b = load_b(&Bts[tc * 16][0], 1, 32, lane);
      acc[u] = wmma_bf16(a, b, acc[u]);
    }
    __syncthreads();
  }
#pragma unroll
  for (int u = 0; u < 2; ++u) {
    const int ti = wave * 2 + u, tr = ti >> 2, tc = ti & 3;
#pragma unroll
    for (int i = 0; i < 8; ++i) {
      const int m = m0 + tr * 16 + lh * 8 + i;
      const int n = n0 + tc * 16 + ln;
      float v = acc[u][i];
      if (ACT == 1) v = siluf(v);
      store_out(&C[(size_t)m * N + n], v);
    }
  }
}

// ---------------- small projections: beta, g ----------------
__global__ void beta_g_kernel(const float* __restrict__ hs, const float* __restrict__ Wb,
                              const float* __restrict__ Wa,
                              const float* __restrict__ A_log,
                              const float* __restrict__ dtb, float* __restrict__ betaf,
                              float* __restrict__ gfv) {
  int idx = blockIdx.x * blockDim.x + threadIdx.x;
  if (idx >= kT * kH) return;
  const int t = idx / kH, h = idx % kH;
  const float* row = hs + (size_t)t * kHID;
  float ab = 0.f, aa = 0.f;
  for (int i = 0; i < kHID; ++i) {
    const float x = row[i];
    ab += x * Wb[(size_t)i * kH + h];
    aa += x * Wa[(size_t)i * kH + h];
  }
  betaf[idx] = 1.f / (1.f + __expf(-ab));
  const float z = aa + dtb[h];
  const float sp = (z > 20.f) ? z : log1pf(__expf(z));
  gfv[idx] = -__expf(A_log[h]) * sp;
}

// ---------------- dynamic causal depthwise conv ----------------
__global__ void conv_kernel(const float* __restrict__ vpre, const float* __restrict__ w,
                            float* __restrict__ vf) {
  int idx = blockIdx.x * blockDim.x + threadIdx.x;
  if (idx >= kT * kVAL) return;
  const int t = idx / kVAL, d = idx % kVAL;
  float acc = 0.f;
#pragma unroll
  for (int kk = 0; kk < kCONV; ++kk) {
    const int ts = t + kk - (kCONV - 1);
    const float vv = (ts >= 0) ? vpre[(size_t)ts * kVAL + d] : 0.f;
    acc += vv * w[(size_t)t * kWCOLS + d * kCONV + kk];
  }
  vf[idx] = siluf(acc);
}

// ---------------- chunked gated delta rule (WMMA) ----------------
// Grid: (H, 2). blockIdx.y selects a 64-wide DV half; each block owns a
// 128(DK) x 64(DV) slice of the per-head state in WMMA C-fragments.
struct ScanShared {
  alignas(16) bf16 Sbt[64][128];   // S mirror, transposed [dv][dk]      16 KB
  alignas(16) bf16 Kb[32][128];    // chunk keys [t][dk]                  8 KB
  alignas(16) bf16 Kbt[128][32];   // chunk keys transposed [dk][t]       8 KB
  alignas(16) float KS0[32][64];   // K@S0, reused for Q@S0               8 KB
  alignas(16) float Mm[32][32];    // substitution matrix                 4 KB
  alignas(16) bf16 Attnb[32][32];  // masked scaled Q@K^T                 2 KB
  alignas(16) bf16 Ubt[64][32];    // u transposed [dv][t]                4 KB
  alignas(16) float Uf[32][64];    // u (f32) [t][dv]                     8 KB
  float Gc[32];                    // in-chunk cumulative log decay
  float Bt_[32];                   // beta
  float gam[32];                   // exp(Gc)
};

__global__ __launch_bounds__(256) void scan_kernel(
    const bf16* __restrict__ qg, const bf16* __restrict__ kg,
    const float* __restrict__ vg, const float* __restrict__ betaf,
    const float* __restrict__ gfv, float* __restrict__ og) {
  const int h = blockIdx.x;
  const int hv = blockIdx.y;  // DV half
  const int tid = threadIdx.x, lane = tid & 31, wave = tid >> 5;
  const int ln = lane & 15, lh = lane >> 4;
  __shared__ ScanShared sh;

  // state slice: wave w owns DK rows [16w,16w+16), 4 column tiles (64 dv)
  v8f S[4];
#pragma unroll
  for (int i = 0; i < 4; ++i) S[i] = vzero8();

  for (int c = 0; c < kT / kCHUNK; ++c) {
    const int t0 = c * kCHUNK;
    // ---- stage K chunk (plain + transposed) + decays ----
    {
      const int r = tid >> 3, cg = (tid & 7) << 4;
      const bf16* src = kg + (size_t)(t0 + r) * kKEY + h * kDK + cg;
      uint4 x0 = *(const uint4*)src;
      uint4 x1 = *(const uint4*)(src + 8);
      *(uint4*)&sh.Kb[r][cg] = x0;
      *(uint4*)&sh.Kb[r][cg + 8] = x1;
      const bf16* q0 = (const bf16*)&x0;
      const bf16* q1 = (const bf16*)&x1;
#pragma unroll
      for (int e = 0; e < 8; ++e) {
        sh.Kbt[cg + e][r]     = q0[e];
        sh.Kbt[cg + 8 + e][r] = q1[e];
      }
    }
    if (tid < 32) {
      sh.Bt_[tid] = betaf[(size_t)(t0 + tid) * kH + h];
      sh.gam[tid] = gfv[(size_t)(t0 + tid) * kH + h];  // raw g for now
    }
    __syncthreads();
    if (tid == 0) {
      float s = 0.f;
      for (int i = 0; i < 32; ++i) { s += sh.gam[i]; sh.Gc[i] = s; }
    }
    __syncthreads();
    if (tid < 32) sh.gam[tid] = __expf(sh.Gc[tid]);
    __syncthreads();

    // ---- Gram -> Mm (waves 0..3), Q@K^T -> Attnb (waves 4..7) ----
    if (wave < 4) {
      const int tr = wave >> 1, tc = wave & 1;
      v8f acc = vzero8();
#pragma unroll
      for (int ks = 0; ks < kDK; ks += 32) {
        v16bf a = load_a(&sh.Kb[tr * 16][ks], 128, 1, lane);
        v16bf b = load_b(&sh.Kb[tc * 16][ks], 1, 128, lane);  // K^T
        acc = wmma_bf16(a, b, acc);
      }
#pragma unroll
      for (int i = 0; i < 8; ++i) {
        const int t = tr * 16 + lh * 8 + i, s = tc * 16 + ln;
        const float dec = __expf(sh.Gc[t] - sh.Gc[s]);     // hoisted: branchless mask
        const float mv = sh.Bt_[t] * dec * acc[i];
        sh.Mm[t][s] = (s < t) ? mv : 0.f;
      }
    } else {
      const int w4 = wave - 4, tr = w4 >> 1, tc = w4 & 1;
      v8f acc = vzero8();
#pragma unroll
      for (int ks = 0; ks < kDK; ks += 32) {
        v16bf a =
            load_a(qg + (size_t)(t0 + tr * 16) * kKEY + h * kDK + ks, kKEY, 1, lane);
        v16bf b = load_b(&sh.Kb[tc * 16][ks], 1, 128, lane);  // K^T
        acc = wmma_bf16(a, b, acc);
      }
#pragma unroll
      for (int i = 0; i < 8; ++i) {
        const int t = tr * 16 + lh * 8 + i, s = tc * 16 + ln;
        const float dec = __expf(sh.Gc[t] - sh.Gc[s]);
        const float av = SCALE_QK * dec * acc[i];
        sh.Attnb[t][s] = f2bf((s <= t) ? av : 0.f);
      }
    }
    __syncthreads();
    const float GcC = sh.Gc[31];
    const float gamC = sh.gam[31];

    // ---- dump S to bf16 transposed LDS mirror ----
#pragma unroll
    for (int tci = 0; tci < 4; ++tci) {
      v8f f = S[tci];
      bf16 tmp[8];
#pragma unroll
      for (int i = 0; i < 8; ++i) tmp[i] = f2bf(f[i]);
      *(uint4*)&sh.Sbt[tci * 16 + ln][16 * wave + lh * 8] = *(const uint4*)tmp;
    }
    __syncthreads();

    // ---- K @ S0 ----
    {
      const int tr = wave >> 2, tcl = wave & 3;
      v8f acc = vzero8();
#pragma unroll
      for (int ks = 0; ks < kDK; ks += 32) {
        v16bf a = load_a(&sh.Kb[tr * 16][ks], 128, 1, lane);
        v16bf b = load_b(&sh.Sbt[tcl * 16][ks], 1, 128, lane);
        acc = wmma_bf16(a, b, acc);
      }
#pragma unroll
      for (int i = 0; i < 8; ++i)
        sh.KS0[tr * 16 + lh * 8 + i][tcl * 16 + ln] = acc[i];
    }
    __syncthreads();

    // ---- forward substitution: u_t = beta_t(v_t - gam_t K S0) - sum M[t][s] u_s ----
    if (tid < 64) {
      const int d = tid;
      const size_t vbase = (size_t)t0 * kVAL + h * kDV + hv * 64 + d;
      for (int t = 0; t < 32; ++t) {
        float accu =
            sh.Bt_[t] * (vg[vbase + (size_t)t * kVAL] - sh.gam[t] * sh.KS0[t][d]);
        for (int s = 0; s < t; ++s) accu -= sh.Mm[t][s] * sh.Uf[s][d];
        sh.Uf[t][d] = accu;
        sh.Ubt[d][t] = f2bf(accu);
      }
    }
    __syncthreads();

    // ---- Q @ S0 (reuse KS0 buffer) ----
    {
      const int tr = wave >> 2, tcl = wave & 3;
      v8f acc = vzero8();
#pragma unroll
      for (int ks = 0; ks < kDK; ks += 32) {
        v16bf a =
            load_a(qg + (size_t)(t0 + tr * 16) * kKEY + h * kDK + ks, kKEY, 1, lane);
        v16bf b = load_b(&sh.Sbt[tcl * 16][ks], 1, 128, lane);
        acc = wmma_bf16(a, b, acc);
      }
#pragma unroll
      for (int i = 0; i < 8; ++i)
        sh.KS0[tr * 16 + lh * 8 + i][tcl * 16 + ln] = acc[i];
    }
    __syncthreads();

    // ---- output: o = Attn @ U + scale * gam_t * (Q S0) ----
    {
      const int tr = wave >> 2, tcl = wave & 3;
      v16bf a = load_a(&sh.Attnb[tr * 16][0], 32, 1, lane);
      v16bf b = load_b(&sh.Ubt[tcl * 16][0], 1, 32, lane);
      v8f acc = wmma_bf16(a, b, vzero8());
#pragma unroll
      for (int i = 0; i < 8; ++i) {
        const int t = tr * 16 + lh * 8 + i, dl = tcl * 16 + ln;
        og[(size_t)(t0 + t) * kVAL + h * kDV + hv * 64 + dl] =
            acc[i] + SCALE_QK * sh.gam[t] * sh.KS0[t][dl];
      }
    }

    // ---- state update: S = gamC*S + (exp(GcC-Gc_s) k_s)^T @ U ----
    {
      v16bf aK;  // A[m=kk][s] = Kbt[16w+m][s] * exp(GcC - Gc[s])
#pragma unroll
      for (int i = 0; i < 8; ++i) {
        const int kb = ((i >> 2) << 4) + (lh << 3) + ((i & 3) << 1);
#pragma unroll
        for (int j = 0; j < 2; ++j) {
          const int s = kb + j;
          aK[2 * i + j] =
              f2bf(bf2f(sh.Kbt[16 * wave + ln][s]) * __expf(GcC - sh.Gc[s]));
        }
      }
#pragma unroll
      for (int tci = 0; tci < 4; ++tci) {
        v8f cacc;
#pragma unroll
        for (int i = 0; i < 8; ++i) cacc[i] = S[tci][i] * gamC;
        v16bf b = load_b(&sh.Ubt[tci * 16][0], 1, 32, lane);
        S[tci] = wmma_bf16(aK, b, cacc);
      }
    }
    __syncthreads();
  }
}

// ---------------- gated RMSNorm ----------------
__global__ __launch_bounds__(128) void norm_kernel(const float* __restrict__ o,
                                                   const float* __restrict__ gate,
                                                   const float* __restrict__ nw,
                                                   bf16* __restrict__ onb) {
  const int row = blockIdx.x;  // t*H + h
  const int t = row / kH, h = row % kH;
  const int d = threadIdx.x;
  const size_t idx = (size_t)t * kVAL + h * kDV + d;
  const float x = o[idx];
  __shared__ float red[128];
  red[d] = x * x;
  __syncthreads();
  for (int s = 64; s > 0; s >>= 1) {
    if (d < s) red[d] += red[d + s];
    __syncthreads();
  }
  const float mean = red[0] * (1.f / kDV);
  const float gv = gate[idx];
  const float val = x * rsqrtf(mean + 1e-5f) * nw[d] * (gv / (1.f + __expf(-gv)));
  onb[idx] = f2bf(val);
}

// ---------------- launcher ----------------
extern "C" void kernel_launch(void* const* d_in, const int* in_sizes, int n_in,
                              void* d_out, int out_size, void* d_ws, size_t ws_size,
                              hipStream_t stream) {
  const float* hs    = (const float*)d_in[0];
  const float* Wq    = (const float*)d_in[1];
  const float* Wk    = (const float*)d_in[2];
  const float* Wv    = (const float*)d_in[3];
  const float* Wb    = (const float*)d_in[4];
  const float* Wa    = (const float*)d_in[5];
  const float* A_log = (const float*)d_in[6];
  const float* dtb   = (const float*)d_in[7];
  const float* Wg1   = (const float*)d_in[8];
  const float* Wg2   = (const float*)d_in[9];
  const float* Wgate = (const float*)d_in[10];
  const float* nw    = (const float*)d_in[11];
  const float* Wo    = (const float*)d_in[12];
  float* out = (float*)d_out;

  char* base = (char*)d_ws;
  size_t off = 0;
  auto alloc = [&](size_t bytes) -> void* {
    void* p = base + off;
    off = (off + bytes + 255) & ~(size_t)255;
    return p;
  };
  bf16* hsb    = (bf16*)alloc((size_t)kT * kHID * 2);
  bf16* wqb    = (bf16*)alloc((size_t)kHID * kKEY * 2);
  bf16* wkb    = (bf16*)alloc((size_t)kHID * kKEY * 2);
  bf16* wvb    = (bf16*)alloc((size_t)kHID * kVAL * 2);
  bf16* wg1b   = (bf16*)alloc((size_t)kHID * kGMID * 2);
  bf16* wg2b   = (bf16*)alloc((size_t)kGMID * kWCOLS * 2);
  bf16* wgateb = (bf16*)alloc((size_t)kHID * kVAL * 2);
  bf16* wob    = (bf16*)alloc((size_t)kVAL * kHID * 2);
  bf16* qb     = (bf16*)alloc((size_t)kT * kKEY * 2);
  bf16* kb     = (bf16*)alloc((size_t)kT * kKEY * 2);
  float* vpre  = (float*)alloc((size_t)kT * kVAL * 4);
  float* gatef = (float*)alloc((size_t)kT * kVAL * 4);
  bf16* gmidb  = (bf16*)alloc((size_t)kT * kGMID * 2);
  float* wconv = (float*)alloc((size_t)kT * kWCOLS * 4);
  float* vf    = (float*)alloc((size_t)kT * kVAL * 4);
  float* betaf = (float*)alloc((size_t)kT * kH * 4);
  float* gfv   = (float*)alloc((size_t)kT * kH * 4);
  float* og    = (float*)alloc((size_t)kT * kVAL * 4);
  bf16* onb    = (bf16*)alloc((size_t)kT * kVAL * 2);

  auto cvt = [&](const float* src, bf16* dst, size_t n) {
    int n4 = (int)(n / 4);
    cvt_kernel<<<dim3((unsigned)((n4 + 255) / 256)), 256, 0, stream>>>(src, dst, n4);
  };
  cvt(hs, hsb, (size_t)kT * kHID);
  cvt(Wq, wqb, (size_t)kHID * kKEY);
  cvt(Wk, wkb, (size_t)kHID * kKEY);
  cvt(Wv, wvb, (size_t)kHID * kVAL);
  cvt(Wg1, wg1b, (size_t)kHID * kGMID);
  cvt(Wg2, wg2b, (size_t)kGMID * kWCOLS);
  cvt(Wgate, wgateb, (size_t)kHID * kVAL);
  cvt(Wo, wob, (size_t)kVAL * kHID);

  // projections (WMMA GEMMs)
  gemm128_kernel<1, bf16><<<dim3(kKEY / 128, kT / 128), 256, 0, stream>>>(
      hsb, wqb, qb, kT, kKEY, kHID);
  gemm128_kernel<1, bf16><<<dim3(kKEY / 128, kT / 128), 256, 0, stream>>>(
      hsb, wkb, kb, kT, kKEY, kHID);
  gemm128_kernel<0, float><<<dim3(kVAL / 128, kT / 128), 256, 0, stream>>>(
      hsb, wvb, vpre, kT, kVAL, kHID);
  gemm128_kernel<0, float><<<dim3(kVAL / 128, kT / 128), 256, 0, stream>>>(
      hsb, wgateb, gatef, kT, kVAL, kHID);
  gemm64_kernel<1, bf16><<<dim3(kGMID / 64, kT / 64), 256, 0, stream>>>(
      hsb, wg1b, gmidb, kT, kGMID, kHID);
  gemm128_kernel<0, float><<<dim3(kWCOLS / 128, kT / 128), 256, 0, stream>>>(
      gmidb, wg2b, wconv, kT, kWCOLS, kGMID);

  // beta / g
  beta_g_kernel<<<dim3((kT * kH + 255) / 256), 256, 0, stream>>>(hs, Wb, Wa, A_log, dtb,
                                                                 betaf, gfv);
  // dynamic conv -> v
  conv_kernel<<<dim3((kT * kVAL + 255) / 256), 256, 0, stream>>>(vpre, wconv, vf);

  // chunked gated delta rule (split DV halves across blockIdx.y)
  scan_kernel<<<dim3(kH, 2), 256, 0, stream>>>(qb, kb, vf, betaf, gfv, og);

  // gated RMSNorm
  norm_kernel<<<dim3(kT * kH), 128, 0, stream>>>(og, gatef, nw, onb);

  // output projection
  gemm128_kernel<0, float><<<dim3(kHID / 128, kT / 128), 256, 0, stream>>>(
      onb, wob, out, kT, kHID, kVAL);
}
